// KernelFilter_62852551410132
// MI455X (gfx1250) — matrix operand order repaired and checked by reading, not compiled, not run on hardware
//
#include <hip/hip_runtime.h>
#include <hip/hip_bf16.h>

// ---------------------------------------------------------------------------
// Types
// ---------------------------------------------------------------------------
typedef __bf16        v16bf __attribute__((ext_vector_type(16)));
typedef float         v8f   __attribute__((ext_vector_type(8)));
typedef unsigned int  u32x4 __attribute__((ext_vector_type(4)));

#define B_    8
#define C_    512
#define H_    64
#define W_    64
#define INNER 32
#define HP    66
#define WP    66

// Workspace layout (bytes, all 1 KB aligned)
static constexpr size_t OFF_CPAD = 0;
static constexpr size_t SZ_CPAD  = (size_t)B_ * HP * WP * C_ * 2;   // 35,684,352
static constexpr size_t OFF_WD   = OFF_CPAD + SZ_CPAD;
static constexpr size_t SZ_WPK   = (size_t)288 * 512 * 2;           // 294,912
static constexpr size_t OFF_WU   = OFF_WD + SZ_WPK;
static constexpr size_t OFF_F1   = OFF_WU + SZ_WPK;
static constexpr size_t SZ_FPK   = (size_t)16 * 512 * 2;            // 16,384
static constexpr size_t OFF_F2   = OFF_F1 + SZ_FPK;
static constexpr size_t OFF_C2   = OFF_F2 + SZ_FPK;
static constexpr size_t SZ_C2    = (size_t)B_ * HP * WP * INNER * 2;

// ---------------------------------------------------------------------------
// Helpers
// ---------------------------------------------------------------------------
__device__ __forceinline__ __bf16 f2bf(float f) {
  unsigned u = __builtin_bit_cast(unsigned, f);
  unsigned r = (u + 0x7FFFu + ((u >> 16) & 1u)) >> 16;   // RNE
  unsigned short s = (unsigned short)r;
  return __builtin_bit_cast(__bf16, s);
}

// 32 contiguous bytes (16 bf16 halves) -> one WMMA operand register block
__device__ __forceinline__ v16bf load_frag32(const __bf16* p) {
  struct Q { u32x4 a, b; } q;
  q.a = *(const u32x4*)(p);
  q.b = *(const u32x4*)(p + 8);
  return __builtin_bit_cast(v16bf, q);
}

__device__ __forceinline__ v8f wmma_bf16(v16bf a, v16bf b, v8f c) {
  return __builtin_amdgcn_wmma_f32_16x16x32_bf16(
      /*neg_a=*/false, a, /*neg_b=*/false, b,
      /*c_mod=*/(short)0, c, /*reuse_a=*/false, /*reuse_b=*/false);
}

// K index within a 32-wide K-tile for (lane, half) of a bf16 operand:
//   klocal = ((h>>3)<<4) | ((lane>>4)<<3) | (h&7)
// Matches 16-bit A layout (ISA 7.12.2); B fragments are packed with the same
// mapping (M<->N swapped) by the weight packers below.

// D-tile pair (N=0..15 in acc0, N=16..31 in acc1) -> LDS as bf16 [m][ch]
__device__ __forceinline__ void d_to_lds(__bf16* ldsW, v8f a0, v8f a1, int lane) {
  const int ch = lane & 15;
  const int mB = (lane >> 4) * 8;
#pragma unroll
  for (int r = 0; r < 8; ++r) {
    ldsW[(mB + r) * 32 + ch]      = f2bf(a0[r]);
    ldsW[(mB + r) * 32 + ch + 16] = f2bf(a1[r]);
  }
}

// LDS [m][ch] (16x32 bf16) -> A fragment (M=16, K=32)
__device__ __forceinline__ v16bf afrag_from_lds(const __bf16* ldsW, int lane) {
  const int m  = lane & 15;
  const int co = (lane >> 4) * 8;
  struct Q { u32x4 a, b; } q;
  q.a = *(const u32x4*)(ldsW + m * 32 + co);
  q.b = *(const u32x4*)(ldsW + m * 32 + co + 16);
  return __builtin_bit_cast(v16bf, q);
}

__device__ __forceinline__ v8f leaky(v8f d) {
#pragma unroll
  for (int r = 0; r < 8; ++r) {
    float v = d[r];
    d[r] = v > 0.f ? v : 0.2f * v;
  }
  return d;
}

// ---------------------------------------------------------------------------
// Packing kernels (run each call; cheap vs. the 128 MB main traffic)
// ---------------------------------------------------------------------------
// content [B,C,H,W] f32 -> zero-padded channels-last bf16 [B,66,66,C]
__global__ void k_pack_content(const float* __restrict__ src, __bf16* __restrict__ dst) {
  const int total = B_ * HP * WP * C_;
  for (int idx = blockIdx.x * blockDim.x + threadIdx.x; idx < total;
       idx += gridDim.x * blockDim.x) {
    int c  = idx & (C_ - 1);
    int p  = idx >> 9;             // b*66*66 + yy*66 + xx
    int xx = p % WP;
    int yy = (p / WP) % HP;
    int b  = p / (WP * HP);
    float v = 0.f;
    if (xx >= 1 && xx <= W_ && yy >= 1 && yy <= H_)
      v = src[((b * C_ + c) * H_ + (yy - 1)) * W_ + (xx - 1)];
    dst[idx] = f2bf(v);
  }
}

// Wd [32,512,3,3] -> fragment-linear: frag = kt*2+nt, kt = tap*16+cblk
__global__ void k_pack_wd(const float* __restrict__ src, __bf16* __restrict__ dst) {
  const int total = 288 * 512;
  for (int idx = blockIdx.x * blockDim.x + threadIdx.x; idx < total;
       idx += gridDim.x * blockDim.x) {
    int h = idx & 15, lane = (idx >> 4) & 31, frag = idx >> 9;
    int kt = frag >> 1, nt = frag & 1;
    int klocal = ((h >> 3) << 4) | ((lane >> 4) << 3) | (h & 7);
    int tap = kt >> 4, cblk = kt & 15;
    int c = cblk * 32 + klocal;
    int o = nt * 16 + (lane & 15);
    dst[idx] = f2bf(src[(o * C_ + c) * 9 + tap]);
  }
}

// Wu [512,32,3,3] -> fragment-linear: frag = nt*9 + tap
__global__ void k_pack_wu(const float* __restrict__ src, __bf16* __restrict__ dst) {
  const int total = 288 * 512;
  for (int idx = blockIdx.x * blockDim.x + threadIdx.x; idx < total;
       idx += gridDim.x * blockDim.x) {
    int h = idx & 15, lane = (idx >> 4) & 31, frag = idx >> 9;
    int nt = frag / 9, tap = frag % 9;
    int klocal = ((h >> 3) << 4) | ((lane >> 4) << 3) | (h & 7);
    int o = nt * 16 + (lane & 15);
    dst[idx] = f2bf(src[(o * INNER + klocal) * 9 + tap]);
  }
}

// filter [B,32,32,1] -> per-batch B fragments (K=i, N=o): frag = b*2+nt
__global__ void k_pack_filt(const float* __restrict__ src, __bf16* __restrict__ dst) {
  const int total = 16 * 512;
  for (int idx = blockIdx.x * blockDim.x + threadIdx.x; idx < total;
       idx += gridDim.x * blockDim.x) {
    int h = idx & 15, lane = (idx >> 4) & 31, frag = idx >> 9;
    int b = frag >> 1, nt = frag & 1;
    int klocal = ((h >> 3) << 4) | ((lane >> 4) << 3) | (h & 7);
    int o = nt * 16 + (lane & 15);
    dst[idx] = f2bf(src[(b * INNER + o) * INNER + klocal]);
  }
}

__global__ void k_zero_u16(unsigned short* __restrict__ p, int n) {
  for (int i = blockIdx.x * blockDim.x + threadIdx.x; i < n;
       i += gridDim.x * blockDim.x)
    p[i] = 0;
}

// ---------------------------------------------------------------------------
// Kernel 1: down-conv (512->32) + bias + filter1 + LeakyReLU + filter2
//           -> padded bf16 c2 [B,66,66,32]
// One wave per M-tile (16 pixels of one row). K loop: 9 taps x 16 c-blocks.
// ---------------------------------------------------------------------------
__global__ void __launch_bounds__(256) k_down(
    const __bf16* __restrict__ cpad, const __bf16* __restrict__ wdp,
    const __bf16* __restrict__ f1p,  const __bf16* __restrict__ f2p,
    const float*  __restrict__ bd,   __bf16* __restrict__ c2) {
  __shared__ __align__(16) __bf16 lds[8 * 512];
  const int lane = threadIdx.x & 31;
  const int wv   = threadIdx.x >> 5;
  const int mt   = blockIdx.x * 8 + wv;           // 0..2047
  const int x0   = (mt & 3) * 16;
  const int y    = (mt >> 2) & 63;
  const int b    = mt >> 8;
  __bf16* ldsW = lds + wv * 512;
  const int col = x0 + (lane & 15);               // pixel column for this lane
  const int co  = (lane >> 4) * 8;                // K sub-offset per A layout

  // Per-batch filter fragments, kept in registers for the whole tile.
  const v16bf f1b0 = load_frag32(f1p + (size_t)(b * 2 + 0) * 512 + lane * 16);
  const v16bf f1b1 = load_frag32(f1p + (size_t)(b * 2 + 1) * 512 + lane * 16);
  const v16bf f2b0 = load_frag32(f2p + (size_t)(b * 2 + 0) * 512 + lane * 16);
  const v16bf f2b1 = load_frag32(f2p + (size_t)(b * 2 + 1) * 512 + lane * 16);

  v8f acc0 = {}; v8f acc1 = {};
#pragma unroll
  for (int ky = 0; ky < 3; ++ky) {
#pragma unroll
    for (int kx = 0; kx < 3; ++kx) {
      const __bf16* rowp =
          cpad + ((size_t)((b * HP + (y + ky)) * WP + (col + kx))) * C_ + co;
      const int tap = ky * 3 + kx;
#pragma unroll 4
      for (int cblk = 0; cblk < 16; ++cblk) {
        struct { u32x4 a, b; } qa;
        qa.a = *(const u32x4*)(rowp + cblk * 32);
        qa.b = *(const u32x4*)(rowp + cblk * 32 + 16);
        v16bf a  = __builtin_bit_cast(v16bf, qa);
        const __bf16* wb = wdp + (size_t)((tap * 16 + cblk) * 2) * 512 + lane * 16;
        v16bf b0 = load_frag32(wb);
        v16bf b1 = load_frag32(wb + 512);
        acc0 = wmma_bf16(a, b0, acc0);
        acc1 = wmma_bf16(a, b1, acc1);
      }
    }
  }

  // bias (channel = lane&15 in D layout)
  const float biasA = bd[lane & 15];
  const float biasB = bd[16 + (lane & 15)];
#pragma unroll
  for (int r = 0; r < 8; ++r) { acc0[r] += biasA; acc1[r] += biasB; }

  // filter1 (32x32 channel matmul) via LDS transpose + WMMA
  d_to_lds(ldsW, acc0, acc1, lane);
  v16bf a1 = afrag_from_lds(ldsW, lane);
  v8f z = {};
  v8f d0 = leaky(wmma_bf16(a1, f1b0, z));
  v8f d1 = leaky(wmma_bf16(a1, f1b1, z));

  // filter2
  d_to_lds(ldsW, d0, d1, lane);
  v16bf a2 = afrag_from_lds(ldsW, lane);
  v8f e0 = wmma_bf16(a2, f2b0, z);
  v8f e1 = wmma_bf16(a2, f2b1, z);

  // write padded channels-last bf16 (coalesced 32B/lane via LDS)
  d_to_lds(ldsW, e0, e1, lane);
  const int m = lane & 15, seg = lane >> 4;
  const u32x4* s = (const u32x4*)(ldsW + m * 32 + seg * 16);
  u32x4* d = (u32x4*)(c2 + ((size_t)((b * HP + (y + 1)) * WP + (x0 + 1 + m))) * INNER +
                      seg * 16);
  d[0] = s[0];
  d[1] = s[1];
}

// ---------------------------------------------------------------------------
// Kernel 2: up-conv (32->512) + bias + residual -> out f32 [B,512,64,64]
// K = 9 taps x 32ch. A fragments (9) loaded once, reused across 32 N-tiles.
// ---------------------------------------------------------------------------
__global__ void __launch_bounds__(256) k_up(
    const __bf16* __restrict__ c2, const __bf16* __restrict__ wup,
    const float* __restrict__ bu, const float* __restrict__ content,
    float* __restrict__ out) {
  const int lane = threadIdx.x & 31;
  const int wv   = threadIdx.x >> 5;
  const int mt   = blockIdx.x * 8 + wv;
  const int x0   = (mt & 3) * 16;
  const int y    = (mt >> 2) & 63;
  const int b    = mt >> 8;
  const int col  = x0 + (lane & 15);
  const int co   = (lane >> 4) * 8;
  const int seg  = lane >> 4;

  v16bf A[9];
#pragma unroll
  for (int t = 0; t < 9; ++t) {
    const int ky = t / 3, kx = t % 3;
    const __bf16* pa =
        c2 + ((size_t)((b * HP + (y + ky)) * WP + (col + kx))) * INNER + co;
    struct { u32x4 a, b; } qa;
    qa.a = *(const u32x4*)(pa);
    qa.b = *(const u32x4*)(pa + 16);
    A[t] = __builtin_bit_cast(v16bf, qa);
  }

#pragma unroll 1
  for (int nt = 0; nt < 32; ++nt) {
    v8f acc = {};
#pragma unroll
    for (int t = 0; t < 9; ++t) {
      v16bf bw = load_frag32(wup + (size_t)(nt * 9 + t) * 512 + lane * 16);
      acc = wmma_bf16(A[t], bw, acc);
    }
    const int ch   = nt * 16 + (lane & 15);
    const float bb = bu[ch];
    const size_t base =
        (((size_t)(b * C_ + ch) * H_ + y) * W_) + x0 + seg * 8;
    float4 r0 = *(const float4*)(content + base);
    float4 r1 = *(const float4*)(content + base + 4);
    float4 o0, o1;
    o0.x = r0.x + acc[0] + bb; o0.y = r0.y + acc[1] + bb;
    o0.z = r0.z + acc[2] + bb; o0.w = r0.w + acc[3] + bb;
    o1.x = r1.x + acc[4] + bb; o1.y = r1.y + acc[5] + bb;
    o1.z = r1.z + acc[6] + bb; o1.w = r1.w + acc[7] + bb;
    *(float4*)(out + base)     = o0;
    *(float4*)(out + base + 4) = o1;
  }
}

// ---------------------------------------------------------------------------
// Launch
// ---------------------------------------------------------------------------
extern "C" void kernel_launch(void* const* d_in, const int* in_sizes, int n_in,
                              void* d_out, int out_size, void* d_ws, size_t ws_size,
                              hipStream_t stream) {
  const float* content = (const float*)d_in[0];
  const float* filter1 = (const float*)d_in[1];
  const float* filter2 = (const float*)d_in[2];
  const float* Wd      = (const float*)d_in[3];
  const float* bd      = (const float*)d_in[4];
  const float* Wu      = (const float*)d_in[5];
  const float* bu      = (const float*)d_in[6];
  float* out = (float*)d_out;

  char* ws = (char*)d_ws;
  __bf16* cpad = (__bf16*)(ws + OFF_CPAD);
  __bf16* wdp  = (__bf16*)(ws + OFF_WD);
  __bf16* wup  = (__bf16*)(ws + OFF_WU);
  __bf16* f1p  = (__bf16*)(ws + OFF_F1);
  __bf16* f2p  = (__bf16*)(ws + OFF_F2);
  __bf16* c2   = (__bf16*)(ws + OFF_C2);

  k_pack_content<<<4096, 256, 0, stream>>>(content, cpad);
  k_pack_wd<<<576, 256, 0, stream>>>(Wd, wdp);
  k_pack_wu<<<576, 256, 0, stream>>>(Wu, wup);
  k_pack_filt<<<32, 256, 0, stream>>>(filter1, f1p);
  k_pack_filt<<<32, 256, 0, stream>>>(filter2, f2p);
  k_zero_u16<<<1024, 256, 0, stream>>>((unsigned short*)c2,
                                       (int)(SZ_C2 / 2));
  k_down<<<256, 256, 0, stream>>>(cpad, wdp, f1p, f2p, bd, c2);
  k_up<<<256, 256, 0, stream>>>(c2, wup, bu, content, out);
}
